// Transformer_v0_57690000720307
// MI455X (gfx1250) — matrix operand order repaired and checked by reading, not compiled, not run on hardware
//
#include <hip/hip_runtime.h>

// ---------------- model constants ----------------
#define Bsz      2
#define NFRAMES  7
#define GROUPS   64
#define NFC      64            // feature channels
#define Lt       8             // tokens per frame
#define CPc      128
#define CTc      512
#define Hc       64
#define Wc       112
#define HWc      7168          // 64*112
#define NLc      56            // NFRAMES*Lt
#define NIMG     14            // B*NFRAMES
#define KCONV    576           // 9*64
#define BGc      128           // B*GROUPS

typedef __attribute__((ext_vector_type(16))) __bf16 v16bf;
typedef __attribute__((ext_vector_type(8)))  __bf16 v8bf;
typedef __attribute__((ext_vector_type(8)))  float  v8f;

union bfcast { unsigned short u; __bf16 b; };

static __device__ __forceinline__ __bf16 bits_to_bf(unsigned short u) {
    bfcast c; c.u = u; return c.b;
}
static __device__ __forceinline__ unsigned short float_to_bfbits(float f) {
    union { float f; unsigned u; } x; x.f = f;
    unsigned r = x.u + 0x7FFFu + ((x.u >> 16) & 1u);   // round-to-nearest-even
    return (unsigned short)(r >> 16);
}

// ---------------- weight packing: OIHW f32 -> fragment-ordered bf16 ----------------
// Layout: frag[((nt*9 + tap)*2 + kch)] holds a full wave fragment:
//   [lane 0..31][e 0..15], value = W[k = tap*64 + kch*32 + (e>>3)*16 + (lane>>4)*8 + (e&7)][oc = nt*16 + (lane&15)]
// => each lane's 16 bf16 fragment elements are CONTIGUOUS (one 32-byte load).
__global__ void pack_weights_kernel(const float* __restrict__ w, unsigned short* __restrict__ dst) {
    int idx = blockIdx.x * 256 + threadIdx.x;
    if (idx >= NFC * NFC * 9) return;
    int e    = idx & 15;
    int t2   = idx >> 4;
    int lane = t2 & 31;
    int t3   = t2 >> 5;
    int kch  = t3 & 1;
    int t4   = t3 >> 1;
    int tap  = t4 % 9;
    int nt   = t4 / 9;
    int oc   = nt * 16 + (lane & 15);
    int ic   = kch * 32 + (e >> 3) * 16 + (lane >> 4) * 8 + (e & 7);        // k within tap
    dst[idx] = float_to_bfbits(w[(oc * NFC + ic) * 9 + tap]);
}

// ---------------- conv_first (3->64) + leaky_relu(0.1) ----------------
__global__ void conv_first_kernel(const float* __restrict__ x, const float* __restrict__ w,
                                  const float* __restrict__ b, float* __restrict__ out) {
    int idx = blockIdx.x * 256 + threadIdx.x;
    if (idx >= NIMG * NFC * HWc) return;
    int p   = idx % HWc;
    int oc  = (idx / HWc) % NFC;
    int img = idx / (NFC * HWc);
    int y0 = p / Wc, x0 = p % Wc;
    float acc = b[oc];
    for (int ic = 0; ic < 3; ++ic)
        for (int dy = -1; dy <= 1; ++dy) {
            int yy = y0 + dy; if (yy < 0 || yy >= Hc) continue;
            for (int dx = -1; dx <= 1; ++dx) {
                int xx = x0 + dx; if (xx < 0 || xx >= Wc) continue;
                acc += w[((oc * 3 + ic) * 3 + (dy + 1)) * 3 + (dx + 1)] *
                       x[(size_t)(img * 3 + ic) * HWc + yy * Wc + xx];
            }
        }
    out[idx] = acc > 0.f ? acc : 0.1f * acc;
}

// ---------------- WMMA implicit-GEMM conv3x3 64->64 (bf16 x bf16 -> f32) ----------------
// grid = (224, nimg): 14 x-tiles (8 wide) x 16 y-tiles (4 tall). block = 256 = 8 wave32s.
// Block tile: 32 spatial (4x8) x 64 outch. Wave tile: 16 spatial x 16 outch.
// Input patch (6 rows x 10 cols x 64 ch) staged once in LDS as [spatial][ic] bf16 so each
// A fragment = 2x ds_load_b128 per lane; B fragments are contiguous 32B global loads.
template <int INBF>
__global__ void conv3x3_wmma_kernel(const void* __restrict__ in_,
                                    const unsigned short* __restrict__ Bfrag,
                                    const float* __restrict__ bias,
                                    const float* __restrict__ res,
                                    float* __restrict__ out, int act) {
    __shared__ unsigned short s_in[60 * 64];           // [r=6*10 spatial][64 ic], 7.5 KB
    const float*          inf = (const float*)in_;
    const unsigned short* inb = (const unsigned short*)in_;
    const int img  = blockIdx.y;
    const int ty0  = ((int)blockIdx.x / 14) * 4;
    const int tx0  = ((int)blockIdx.x % 14) * 8;
    const int tid  = threadIdx.x;
    const size_t ibase = (size_t)img * NFC * HWc;

    // ---- stage input patch with zero-fill halo (coalesced along x) ----
    for (int idx = tid; idx < 60 * 64; idx += 256) {
        int ic = idx / 60;
        int r  = idx % 60;
        int yy = ty0 + (r / 10) - 1;
        int xx = tx0 + (r % 10) - 1;
        bool ok = (yy >= 0) & (yy < Hc) & (xx >= 0) & (xx < Wc);
        unsigned short u = 0;
        if (ok) {
            size_t gi = ibase + (size_t)ic * HWc + yy * Wc + xx;
            u = INBF ? inb[gi] : float_to_bfbits(inf[gi]);
        }
        s_in[r * 64 + ic] = u;
    }
    __syncthreads();

    const int wv     = tid >> 5;
    const int lane   = tid & 31;
    const int half   = lane >> 4;
    const int lane16 = lane & 15;
    const int mt     = wv >> 2;            // wave M sub-tile (0/1)
    const int nt     = wv & 3;             // wave N sub-tile (0..3)
    const int m      = mt * 16 + lane16;   // A row (spatial) for this lane
    const int ty     = m >> 3, tx = m & 7;
    const int klo    = half * 8;           // A K-octet base for this half-wave
    const v16bf* wp  = (const v16bf*)Bfrag;

    v8f acc = {0.f, 0.f, 0.f, 0.f, 0.f, 0.f, 0.f, 0.f};

#pragma unroll
    for (int tap = 0; tap < 9; ++tap) {
        const int dy = tap / 3, dx = tap % 3;
        const int rsp = (ty + dy) * 10 + (tx + dx);
        if (tap < 8)  // prefetch next tap's weight fragments -> global_prefetch_b8
            __builtin_prefetch(&wp[((size_t)(nt * 9 + tap + 1) * 2) * 32 + lane], 0, 0);
#pragma unroll
        for (int kch = 0; kch < 2; ++kch) {
            // A: 2x 16-byte LDS reads (ic = kc+klo..+7 and kc+klo+16..+23), 16B aligned
            const unsigned short* ap = &s_in[rsp * 64 + kch * 32 + klo];
            v8bf lo = *(const v8bf*)ap;
            v8bf hi = *(const v8bf*)(ap + 16);
            v16bf a = __builtin_shufflevector(lo, hi,
                        0, 1, 2, 3, 4, 5, 6, 7, 8, 9, 10, 11, 12, 13, 14, 15);
            // B: one contiguous 32-byte global load per lane
            v16bf bm = wp[(size_t)((nt * 9 + tap) * 2 + kch) * 32 + lane];
            acc = __builtin_amdgcn_wmma_f32_16x16x32_bf16(
                      false, a, false, bm, (short)0, acc, false, false);
        }
    }

    const int oc = nt * 16 + lane16;       // C/D: N = lane%16
    const float bv = bias[oc];
#pragma unroll
    for (int r = 0; r < 8; ++r) {
        int mm = mt * 16 + r + half * 8;   // C/D: M = r + 8*(lane>=16)
        int pp = (ty0 + (mm >> 3)) * Wc + tx0 + (mm & 7);
        float v = acc[r] + bv;
        if (act == 1) v = fmaxf(v, 0.f);
        size_t oi = ibase + (size_t)oc * HWc + pp;
        if (res) v += res[oi];
        out[oi] = v;
    }
}

// ---------------- tokenizer: tok_coef conv1x1 + softmax over HW ----------------
// grid = (L=8, Bn=14); coefL layout [bn][l][p]
__global__ void tok_coef_kernel(const float* __restrict__ feat, const float* __restrict__ tw,
                                const float* __restrict__ tb, float* __restrict__ coefL) {
    const int l = blockIdx.x, bn = blockIdx.y, tid = threadIdx.x;
    __shared__ float sw[NFC];
    __shared__ float red[256];
    float* row = coefL + (size_t)(bn * Lt + l) * HWc;
    const float* f = feat + (size_t)bn * NFC * HWc;
    if (tid < NFC) sw[tid] = tw[l * NFC + tid];
    __syncthreads();
    float lmax = -1e30f;
    const float bias = tb[l];
    for (int p = tid; p < HWc; p += 256) {
        float acc = bias;
        for (int c = 0; c < NFC; ++c) acc += sw[c] * f[(size_t)c * HWc + p];
        row[p] = acc;
        lmax = fmaxf(lmax, acc);
    }
    red[tid] = lmax; __syncthreads();
    for (int s = 128; s > 0; s >>= 1) { if (tid < s) red[tid] = fmaxf(red[tid], red[tid + s]); __syncthreads(); }
    const float mx = red[0]; __syncthreads();
    float lsum = 0.f;
    for (int p = tid; p < HWc; p += 256) { float e = __expf(row[p] - mx); row[p] = e; lsum += e; }
    red[tid] = lsum; __syncthreads();
    for (int s = 128; s > 0; s >>= 1) { if (tid < s) red[tid] += red[tid + s]; __syncthreads(); }
    const float inv = 1.f / red[0];
    for (int p = tid; p < HWc; p += 256) row[p] *= inv;
}

// value = conv1x1 64->64
__global__ void value_kernel(const float* __restrict__ feat, const float* __restrict__ vw,
                             const float* __restrict__ vb, float* __restrict__ val) {
    int idx = blockIdx.x * 256 + threadIdx.x;
    if (idx >= NIMG * NFC * HWc) return;
    int p = idx % HWc, c = (idx / HWc) % NFC, bn = idx / (NFC * HWc);
    const float* f = feat + (size_t)bn * NFC * HWc;
    const float* w = vw + c * NFC;
    float acc = vb[c];
    for (int ic = 0; ic < NFC; ++ic) acc += w[ic] * f[(size_t)ic * HWc + p];
    val[idx] = acc;
}

// tokens (64xL) = value @ coef
__global__ void tokens_kernel(const float* __restrict__ val, const float* __restrict__ coefL,
                              float* __restrict__ tok64) {
    int idx = blockIdx.x * 256 + threadIdx.x;
    if (idx >= NIMG * NFC * Lt) return;
    int l = idx & 7, c = (idx >> 3) & 63, bn = idx >> 9;
    const float* v  = val   + (size_t)(bn * NFC + c) * HWc;
    const float* co = coefL + (size_t)(bn * Lt + l) * HWc;
    float acc = 0.f;
    for (int p = 0; p < HWc; ++p) acc += v[p] * co[p];
    tok64[idx] = acc;
}

// pos (CPxL) = pos_w @ fix  (resize is identity: DS_H,DS_W == H,W)
__global__ void pos_kernel(const float* __restrict__ coefL, const float* __restrict__ pw,
                           const float* __restrict__ pb, float* __restrict__ pos) {
    int idx = blockIdx.x * 256 + threadIdx.x;
    if (idx >= NIMG * CPc * Lt) return;
    int l = idx & 7, cp = (idx >> 3) & 127, bn = idx >> 10;
    const float* co = coefL + (size_t)(bn * Lt + l) * HWc;
    const float* w  = pw + (size_t)cp * HWc;
    float acc = pb[cp];
    for (int p = 0; p < HWc; ++p) acc += w[p] * co[p];
    pos[idx] = acc;
}

// fuse: cat([tokens;pos]) -> CT    tokens layout [bn][ct][l]
__global__ void fuse_kernel(const float* __restrict__ tok64, const float* __restrict__ pos,
                            const float* __restrict__ fw, const float* __restrict__ fb,
                            float* __restrict__ tokens) {
    int idx = blockIdx.x * 256 + threadIdx.x;
    if (idx >= NIMG * CTc * Lt) return;
    int l = idx & 7, ct = (idx >> 3) & 511, bn = idx >> 12;
    const float* w = fw + ct * (NFC + CPc);
    float acc = fb[ct];
    for (int j = 0; j < NFC; ++j)  acc += w[j]       * tok64[(bn * NFC + j) * Lt + l];
    for (int j = 0; j < CPc; ++j)  acc += w[NFC + j] * pos[(bn * CPc + j) * Lt + l];
    tokens[idx] = acc;
}

// ---------------- static transformer, one block per (group, batch) ----------------
__global__ void static_transformer_kernel(const float* __restrict__ tokens,
        const float* __restrict__ fw, const float* __restrict__ fb,
        const float* __restrict__ kw, const float* __restrict__ kb,
        const float* __restrict__ qw, const float* __restrict__ qb,
        const float* __restrict__ vw, const float* __restrict__ vb,
        const float* __restrict__ lw, const float* __restrict__ lb,
        float* __restrict__ gt) {
    const int g = blockIdx.x, b = blockIdx.y, tid = threadIdx.x;
    __shared__ float s_center[CTc * Lt];
    __shared__ float s_inter [CTc * Lt];   // later reused as kqv
    __shared__ float s_buf   [CTc * Lt];   // k|q, later reused as v
    __shared__ float s_coef  [NLc * Lt];
    __shared__ float s_kq    [Lt * Lt];
    const float* tokB = tokens + (size_t)(b * NFRAMES) * CTc * Lt;

    for (int i = tid; i < CTc * Lt; i += 256) s_center[i] = tokB[3 * CTc * Lt + i];
    __syncthreads();
    // coef = softmax_NL(fw @ center + fb)
    for (int i = tid; i < NLc * Lt; i += 256) {
        int j = i >> 3, l = i & 7;
        const float* wr = fw + (size_t)(g * NLc + j) * CTc;
        float acc = fb[g * NLc + j];
        for (int c = 0; c < CTc; ++c) acc += wr[c] * s_center[c * Lt + l];
        s_coef[i] = acc;
    }
    __syncthreads();
    if (tid < Lt) {
        int l = tid; float mx = -1e30f;
        for (int j = 0; j < NLc; ++j) mx = fmaxf(mx, s_coef[j * Lt + l]);
        float sm = 0.f;
        for (int j = 0; j < NLc; ++j) { float e = __expf(s_coef[j * Lt + l] - mx); s_coef[j * Lt + l] = e; sm += e; }
        float inv = 1.f / sm;
        for (int j = 0; j < NLc; ++j) s_coef[j * Lt + l] *= inv;
    }
    __syncthreads();
    // inter = center + total @ coef
    for (int i = tid; i < CTc * Lt; i += 256) {
        int c = i >> 3, l = i & 7;
        float acc = s_center[i];
        for (int n = 0; n < NFRAMES; ++n) {
            const float* tk = tokB + (size_t)n * CTc * Lt + c * Lt;
            for (int m = 0; m < Lt; ++m) acc += tk[m] * s_coef[(n * Lt + m) * Lt + l];
        }
        s_inter[i] = acc;
    }
    __syncthreads();
    // k (256x8) into s_buf[0:2048), q into s_buf[2048:4096)
    for (int i = tid; i < 4096; i += 256) {
        int hf = i >> 11, r = i & 2047, c = r >> 3, l = r & 7;
        const float* W = (hf ? qw : kw) + (size_t)(g * 256 + c) * CTc;
        float acc = (hf ? qb : kb)[g * 256 + c];
        for (int t = 0; t < CTc; ++t) acc += W[t] * s_inter[t * Lt + l];
        s_buf[hf * 2048 + r] = acc;
    }
    __syncthreads();
    if (tid < 64) {
        int l = tid >> 3, m = tid & 7;
        float acc = 0.f;
        for (int c = 0; c < 256; ++c) acc += s_buf[c * Lt + l] * s_buf[2048 + c * Lt + m];
        s_kq[l * Lt + m] = acc;
    }
    __syncthreads();
    if (tid < Lt) {   // softmax over axis=1 (l) for fixed m
        int m = tid; float mx = -1e30f;
        for (int l = 0; l < Lt; ++l) mx = fmaxf(mx, s_kq[l * Lt + m]);
        float sm = 0.f;
        for (int l = 0; l < Lt; ++l) { float e = __expf(s_kq[l * Lt + m] - mx); s_kq[l * Lt + m] = e; sm += e; }
        float inv = 1.f / sm;
        for (int l = 0; l < Lt; ++l) s_kq[l * Lt + m] *= inv;
    }
    __syncthreads();
    // v = vw @ inter + vb  -> overwrite s_buf
    for (int i = tid; i < CTc * Lt; i += 256) {
        int c = i >> 3, l = i & 7;
        const float* W = vw + (size_t)(g * CTc + c) * CTc;
        float acc = vb[g * CTc + c];
        for (int t = 0; t < CTc; ++t) acc += W[t] * s_inter[t * Lt + l];
        s_buf[i] = acc;
    }
    __syncthreads();
    // kqv = v @ kq + center -> overwrite s_inter
    for (int i = tid; i < CTc * Lt; i += 256) {
        int c = i >> 3, l = i & 7;
        float acc = s_center[i];
        for (int m = 0; m < Lt; ++m) acc += s_buf[c * Lt + m] * s_kq[m * Lt + l];
        s_inter[i] = acc;
    }
    __syncthreads();
    // out = center + lw @ kqv + lb
    float* out = gt + (size_t)(g * Bsz + b) * CTc * Lt;
    for (int i = tid; i < CTc * Lt; i += 256) {
        int c = i >> 3, l = i & 7;
        const float* W = lw + (size_t)(g * CTc + c) * CTc;
        float acc = s_center[i] + lb[g * CTc + c];
        for (int t = 0; t < CTc; ++t) acc += W[t] * s_inter[t * Lt + l];
        out[i] = acc;
    }
}

// ---------------- projector q = proj_q_w @ center_feat ----------------
__global__ void proj_q_kernel(const float* __restrict__ feat, const float* __restrict__ qw,
                              const float* __restrict__ qb, float* __restrict__ qbuf) {
    int idx = blockIdx.x * 256 + threadIdx.x;
    if (idx >= Bsz * NFC * HWc) return;
    int b = idx / (NFC * HWc), c = (idx / HWc) % NFC, p = idx % HWc;
    const float* f = feat + (size_t)(b * NFRAMES + 3) * NFC * HWc;
    const float* w = qw + c * NFC;
    float acc = qb[c];
    for (int ic = 0; ic < NFC; ++ic) acc += w[ic] * f[(size_t)ic * HWc + p];
    qbuf[idx] = acc;
}

// ---------------- projector: proj = feat + v @ softmax_l(q^T k); store bf16 ----------------
__global__ void projector_kernel(const float* __restrict__ gt, const float* __restrict__ qbuf,
                                 const float* __restrict__ feat,
                                 const float* __restrict__ pkw, const float* __restrict__ pkb,
                                 const float* __restrict__ pvw, const float* __restrict__ pvb,
                                 unsigned short* __restrict__ projbf) {
    const int g = blockIdx.x, b = blockIdx.y, tid = threadIdx.x;
    const int bg = b * GROUPS + g;
    const int bq = bg & 1;   // reference's tile-vs-reshape interleave: feat index = (b*G+g) % B
    __shared__ float s_k[NFC * Lt], s_v[NFC * Lt];
    const float* toks = gt + (size_t)(g * Bsz + b) * CTc * Lt;
    for (int i = tid; i < 2 * NFC * Lt; i += 256) {
        int hf = i >> 9, r = i & 511, c = r >> 3, l = r & 7;
        const float* W = (hf ? pvw : pkw) + (size_t)c * CTc;
        float acc = (hf ? pvb : pkb)[c];
        for (int t = 0; t < CTc; ++t) acc += W[t] * toks[t * Lt + l];
        (hf ? s_v : s_k)[r] = acc;
    }
    __syncthreads();
    const float* qb_ = qbuf + (size_t)bq * NFC * HWc;
    const float* fb_ = feat + (size_t)(bq * NFRAMES + 3) * NFC * HWc;
    for (int p = tid; p < HWc; p += 256) {
        float sc[Lt];
        for (int l = 0; l < Lt; ++l) sc[l] = 0.f;
        for (int c = 0; c < NFC; ++c) {
            float qv = qb_[(size_t)c * HWc + p];
            for (int l = 0; l < Lt; ++l) sc[l] += qv * s_k[c * Lt + l];
        }
        float mx = sc[0];
        for (int l = 1; l < Lt; ++l) mx = fmaxf(mx, sc[l]);
        float sm = 0.f;
        for (int l = 0; l < Lt; ++l) { sc[l] = __expf(sc[l] - mx); sm += sc[l]; }
        float inv = 1.f / sm;
        for (int l = 0; l < Lt; ++l) sc[l] *= inv;
        for (int c = 0; c < NFC; ++c) {
            float acc = fb_[(size_t)c * HWc + p];
            for (int l = 0; l < Lt; ++l) acc += s_v[c * Lt + l] * sc[l];
            projbf[((size_t)bg * NFC + c) * HWc + p] = float_to_bfbits(acc);
        }
    }
}

// ---------------- launch ----------------
extern "C" void kernel_launch(void* const* d_in, const int* in_sizes, int n_in,
                              void* d_out, int out_size, void* d_ws, size_t ws_size,
                              hipStream_t stream) {
    (void)in_sizes; (void)n_in; (void)out_size; (void)ws_size;
    const float* x      = (const float*)d_in[0];
    const float* cfw    = (const float*)d_in[1];
    const float* cfb    = (const float*)d_in[2];
    const float* rbw1   = (const float*)d_in[3];
    const float* rbb1   = (const float*)d_in[4];
    const float* rbw2   = (const float*)d_in[5];
    const float* rbb2   = (const float*)d_in[6];
    const float* tokw   = (const float*)d_in[7];
    const float* tokb   = (const float*)d_in[8];
    const float* valw   = (const float*)d_in[9];
    const float* valb   = (const float*)d_in[10];
    const float* posw   = (const float*)d_in[11];
    const float* posbp  = (const float*)d_in[12];
    const float* fusew  = (const float*)d_in[13];
    const float* fuseb  = (const float*)d_in[14];
    const float* trfw   = (const float*)d_in[15];
    const float* trfb   = (const float*)d_in[16];
    const float* trkw   = (const float*)d_in[17];
    const float* trkb   = (const float*)d_in[18];
    const float* trqw   = (const float*)d_in[19];
    const float* trqb   = (const float*)d_in[20];
    const float* trvw   = (const float*)d_in[21];
    const float* trvb   = (const float*)d_in[22];
    const float* trlw   = (const float*)d_in[23];
    const float* trlb   = (const float*)d_in[24];
    const float* pkw    = (const float*)d_in[25];
    const float* pkb    = (const float*)d_in[26];
    const float* pqw    = (const float*)d_in[27];
    const float* pqb    = (const float*)d_in[28];
    const float* pvw    = (const float*)d_in[29];
    const float* pvb    = (const float*)d_in[30];
    const float* c3w    = (const float*)d_in[31];
    const float* c3b    = (const float*)d_in[32];

    // workspace linear allocator
    char* base = (char*)d_ws;
    size_t off = 0;
    auto alloc = [&](size_t bytes) -> char* {
        char* p = base + off;
        off += (bytes + 255) & ~(size_t)255;
        return p;
    };
    float* fA     = (float*)alloc((size_t)NIMG * NFC * HWc * 4);
    float* fT     = (float*)alloc((size_t)NIMG * NFC * HWc * 4);
    float* coefL  = (float*)alloc((size_t)NIMG * Lt * HWc * 4);
    float* val    = (float*)alloc((size_t)NIMG * NFC * HWc * 4);
    float* tok64  = (float*)alloc((size_t)NIMG * NFC * Lt * 4);
    float* posb   = (float*)alloc((size_t)NIMG * CPc * Lt * 4);
    float* tokens = (float*)alloc((size_t)NIMG * CTc * Lt * 4);
    float* gt     = (float*)alloc((size_t)GROUPS * Bsz * CTc * Lt * 4);
    float* qbuf   = (float*)alloc((size_t)Bsz * NFC * HWc * 4);
    unsigned short* projbf = (unsigned short*)alloc((size_t)BGc * NFC * HWc * 2);
    unsigned short* wpack  = (unsigned short*)alloc((size_t)11 * KCONV * NFC * 2);

    const dim3 blk(256);
    const int WSZ = KCONV * NFC;   // 36864 elems per packed conv layer
    // pack all 11 conv weight tensors to fragment-ordered bf16
    for (int i = 0; i < 5; ++i) {
        pack_weights_kernel<<<144, blk, 0, stream>>>(rbw1 + (size_t)i * WSZ, wpack + (size_t)(2 * i) * WSZ);
        pack_weights_kernel<<<144, blk, 0, stream>>>(rbw2 + (size_t)i * WSZ, wpack + (size_t)(2 * i + 1) * WSZ);
    }
    pack_weights_kernel<<<144, blk, 0, stream>>>(c3w, wpack + (size_t)10 * WSZ);

    // conv_first + leaky relu
    conv_first_kernel<<<(NIMG * NFC * HWc + 255) / 256, blk, 0, stream>>>(x, cfw, cfb, fA);

    // 5 residual blocks (WMMA implicit GEMM)
    dim3 gridc(224, NIMG);
    for (int i = 0; i < 5; ++i) {
        conv3x3_wmma_kernel<0><<<gridc, blk, 0, stream>>>(
            fA, wpack + (size_t)(2 * i) * WSZ, rbb1 + i * NFC, nullptr, fT, /*act=relu*/1);
        conv3x3_wmma_kernel<0><<<gridc, blk, 0, stream>>>(
            fT, wpack + (size_t)(2 * i + 1) * WSZ, rbb2 + i * NFC, /*res=*/fA, fA, /*act=*/0);
    }

    // tokenizer
    tok_coef_kernel<<<dim3(Lt, NIMG), blk, 0, stream>>>(fA, tokw, tokb, coefL);
    value_kernel<<<(NIMG * NFC * HWc + 255) / 256, blk, 0, stream>>>(fA, valw, valb, val);
    tokens_kernel<<<(NIMG * NFC * Lt + 255) / 256, blk, 0, stream>>>(val, coefL, tok64);
    pos_kernel<<<(NIMG * CPc * Lt + 255) / 256, blk, 0, stream>>>(coefL, posw, posbp, posb);
    fuse_kernel<<<(NIMG * CTc * Lt + 255) / 256, blk, 0, stream>>>(tok64, posb, fusew, fuseb, tokens);

    // per-group static transformer
    static_transformer_kernel<<<dim3(GROUPS, Bsz), blk, 0, stream>>>(
        tokens, trfw, trfb, trkw, trkb, trqw, trqb, trvw, trvb, trlw, trlb, gt);

    // projector
    proj_q_kernel<<<(Bsz * NFC * HWc + 255) / 256, blk, 0, stream>>>(fA, pqw, pqb, qbuf);
    projector_kernel<<<dim3(GROUPS, Bsz), blk, 0, stream>>>(gt, qbuf, fA, pkw, pkb, pvw, pvb, projbf);

    // final conv3x3 over 128 images (WMMA, bf16 input) -> d_out
    conv3x3_wmma_kernel<1><<<dim3(224, BGc), blk, 0, stream>>>(
        projbf, wpack + (size_t)10 * WSZ, c3b, nullptr, (float*)d_out, /*act=*/0);
}
